// AttentionConv2D_82789789597936
// MI455X (gfx1250) — matrix operand order repaired and checked by reading, not compiled, not run on hardware
//
#include <hip/hip_runtime.h>

// ---------------------------------------------------------------------------
// AttentionConv2D for MI455X (gfx1250): bf16 WMMA for the 4 big GEMMs,
// fp32 VALU for the tiny 9-position softmax attention.
// B=8, CIN=COUT=128, H=W=64, KS=3, NH=4, A=OSH=32, K2=9, pixels=32768.
// ---------------------------------------------------------------------------

typedef __bf16 bf16_t;
typedef __attribute__((ext_vector_type(16))) __bf16 v16bf;
typedef __attribute__((ext_vector_type(8)))  float  v8f;

#define NPIX   32768     // 8*64*64
#define TILE_M 128       // pixels per block
#define NBLK   (NPIX / TILE_M)

// LDS strides (padded to dodge bank conflicts)
#define XS_STR  132      // fp32 rows
#define XB_STR  136      // bf16 rows

// ---------------------------------------------------------------------------
// Kernel 1: pos_k[9][128] = sinusoidal PE(9 x 128) @ Wp + bp  (tiny, 1 block)
// ---------------------------------------------------------------------------
__global__ __launch_bounds__(128) void posk_kernel(
    const float* __restrict__ Wp, const float* __restrict__ bp,
    float* __restrict__ posk)
{
  const int n = threadIdx.x;                    // output column 0..127
  const float c0 = 0.07195578415606393f;        // ln(10000)/128
  for (int p = 0; p < 9; ++p) {
    float s = bp[n];
    for (int i = 0; i < 64; ++i) {
      float div = __expf(-(float)(2 * i) * c0);
      float ang = (float)p * div;
      s += __sinf(ang) * Wp[(2 * i) * 128 + n];
      s += __cosf(ang) * Wp[(2 * i + 1) * 128 + n];
    }
    posk[p * 128 + n] = s;
  }
}

// ---------------------------------------------------------------------------
// Kernel 2: convert+transpose weights fp32 [k][n] -> bf16 [n][k]
// block 0..3 -> Wq, Wk, Wv, Wf
// ---------------------------------------------------------------------------
__global__ __launch_bounds__(256) void wconv_kernel(
    const float* __restrict__ W0, const float* __restrict__ W1,
    const float* __restrict__ W2, const float* __restrict__ W3,
    unsigned short* __restrict__ T0, unsigned short* __restrict__ T1,
    unsigned short* __restrict__ T2, unsigned short* __restrict__ T3)
{
  const float* W = (blockIdx.x == 0) ? W0 : (blockIdx.x == 1) ? W1
                 : (blockIdx.x == 2) ? W2 : W3;
  bf16_t* T = (bf16_t*)((blockIdx.x == 0) ? T0 : (blockIdx.x == 1) ? T1
                      : (blockIdx.x == 2) ? T2 : T3);
  for (int i = 0; i < 64; ++i) {
    int idx = threadIdx.x + i * 256;            // 0..16383
    int n = idx & 127, k = idx >> 7;            // coalesced read over n
    T[n * 128 + k] = (bf16_t)W[k * 128 + n];
  }
}

// ---------------------------------------------------------------------------
// A-fragment loader: 16-bit A matrix 16x32, contiguous K per lane half (same order as B).
// lane = m(0..15) + 16*khalf ; VGPR v holds K = {khalf*16+2v, +1}
// ---------------------------------------------------------------------------
__device__ __forceinline__ v16bf load_a_frag(const bf16_t* __restrict__ rowp,
                                             int khalf)
{
  v16bf f;
#pragma unroll
  for (int v = 0; v < 8; ++v) {
    int kk = khalf * 16 + 2 * v;
    f[2 * v]     = rowp[kk];
    f[2 * v + 1] = rowp[kk + 1];
  }
  return f;
}

// ---------------------------------------------------------------------------
// Kernel 3: LayerNorm + Q/K/V projections (3 x GEMM 128x128x128 per tile)
// block = 256 threads = 8 waves, 128 pixels per block.
// ---------------------------------------------------------------------------
__global__ __launch_bounds__(256) void ln_qkv_kernel(
    const float* __restrict__ x,
    const float* __restrict__ ln_g, const float* __restrict__ ln_b,
    const unsigned short* __restrict__ WqTp,
    const unsigned short* __restrict__ WkTp,
    const unsigned short* __restrict__ WvTp,
    const float* __restrict__ bq, const float* __restrict__ bk,
    const float* __restrict__ bv,
    float* __restrict__ Qbuf, float* __restrict__ Kbuf,
    float* __restrict__ Vbuf)
{
  extern __shared__ char smem[];
  float*  xs  = (float*)smem;                               // [128][XS_STR]
  bf16_t* xnb = (bf16_t*)(smem + TILE_M * XS_STR * 4);      // [128][XB_STR]

  const int tid = threadIdx.x;
  const int p0  = blockIdx.x * TILE_M;
  const int b   = p0 >> 12;

  // --- coalesced gather of channel-major input into [pixel][channel] LDS ---
#pragma unroll 4
  for (int i = 0; i < 64; ++i) {
    int idx = tid + i * 256;                 // 0..16383
    int c = idx >> 7, pix = idx & 127;
    int p = p0 + pix;
    int h = (p >> 6) & 63, w = p & 63;
    xs[pix * XS_STR + c] = x[((b * 128 + c) << 12) + (h << 6) + w];
  }
  __syncthreads();

  // --- LayerNorm (one pixel per thread, 128 threads) -> bf16 LDS ---
  if (tid < 128) {
    float mu = 0.f;
#pragma unroll 8
    for (int c = 0; c < 128; ++c) mu += xs[tid * XS_STR + c];
    mu *= (1.f / 128.f);
    float var = 0.f;
#pragma unroll 8
    for (int c = 0; c < 128; ++c) {
      float d = xs[tid * XS_STR + c] - mu;
      var += d * d;
    }
    var *= (1.f / 128.f);
    float inv = rsqrtf(var + 1e-5f);
#pragma unroll 8
    for (int c = 0; c < 128; ++c) {
      float xv = (xs[tid * XS_STR + c] - mu) * inv * ln_g[c] + ln_b[c];
      xnb[tid * XB_STR + c] = (bf16_t)xv;
    }
  }
  __syncthreads();

  // --- WMMA GEMMs: each wave owns 16 M-rows, loops 3 matrices x 8 N-tiles ---
  const int lane  = tid & 31;
  const int wave  = tid >> 5;
  const int mlo   = lane & 15;
  const int khalf = lane >> 4;
  const int mrow  = wave * 16 + mlo;

  v16bf afrag[4];
#pragma unroll
  for (int kc = 0; kc < 4; ++kc)
    afrag[kc] = load_a_frag(xnb + mrow * XB_STR + kc * 32, khalf);

  const int mstore = wave * 16 + khalf * 8;   // C/D: lanes 16-31 hold M=r+8
#pragma unroll
  for (int mat = 0; mat < 3; ++mat) {
    const bf16_t* WT  = (const bf16_t*)(mat == 0 ? WqTp : (mat == 1 ? WkTp : WvTp));
    const float* bias = (mat == 0 ? bq : (mat == 1 ? bk : bv));
    float* Obuf       = (mat == 0 ? Qbuf : (mat == 1 ? Kbuf : Vbuf));
    for (int nt = 0; nt < 8; ++nt) {
      const int n = nt * 16 + mlo;
      const float bb = bias[n];
      v8f acc = { bb, bb, bb, bb, bb, bb, bb, bb };
#pragma unroll
      for (int kc = 0; kc < 4; ++kc) {
        // B fragment: 16 contiguous bf16 along K from transposed weights
        v16bf bfrag = *(const v16bf*)(WT + n * 128 + kc * 32 + khalf * 16);
        acc = __builtin_amdgcn_wmma_f32_16x16x32_bf16(
            false, afrag[kc], false, bfrag, (short)0, acc, false, false);
      }
#pragma unroll
      for (int r = 0; r < 8; ++r)
        Obuf[(p0 + mstore + r) * 128 + n] = acc[r];
    }
  }
}

// ---------------------------------------------------------------------------
// Kernel 4: window attention (fp32 VALU) + final 128x128 WMMA GEMM + NCHW store
// ---------------------------------------------------------------------------
__global__ __launch_bounds__(256) void attn_final_kernel(
    const float* __restrict__ Qbuf, const float* __restrict__ Kbuf,
    const float* __restrict__ Vbuf, const float* __restrict__ posk,
    const unsigned short* __restrict__ WfTp, const float* __restrict__ bf_,
    float* __restrict__ out)
{
  extern __shared__ char smem[];
  bf16_t* att = (bf16_t*)smem;                                    // [128][XB_STR]
  float*  res = (float*)(smem + TILE_M * XB_STR * 2);             // [128][XS_STR]
  float*  pks = (float*)(smem + TILE_M * XB_STR * 2 + TILE_M * XS_STR * 4); // [9][128]

  const int tid = threadIdx.x;
  const int p0  = blockIdx.x * TILE_M;
  const int b   = p0 >> 12;

  if (tid < 128) {
#pragma unroll
    for (int pp = 0; pp < 9; ++pp) pks[pp * 128 + tid] = posk[pp * 128 + tid];
  }
  __syncthreads();

  // --- attention: 128 pixels x 4 heads = 512 tasks, 2 per thread ---
  for (int t = 0; t < 2; ++t) {
    const int task = tid + t * 256;
    const int pix = task >> 2, head = task & 3;
    const int p = p0 + pix;
    const int h = (p >> 6) & 63, w = p & 63;

    float q[32];
    const float* qp = Qbuf + p * 128 + head * 32;
#pragma unroll
    for (int a = 0; a < 32; ++a) q[a] = qp[a];

    float sc[9];
#pragma unroll
    for (int pp = 0; pp < 9; ++pp) {
      const int di = pp / 3 - 1, dj = pp % 3 - 1;
      const int hh = h + di, wj = w + dj;
      const bool inb = (hh >= 0) && (hh < 64) && (wj >= 0) && (wj < 64);
      const float* kr = Kbuf + ((((b << 6) + hh) << 6) + wj) * 128 + head * 32;
      const float* pk = pks + pp * 128 + head * 32;
      float s = 0.f;
      if (inb) {
#pragma unroll
        for (int a = 0; a < 32; ++a) s += q[a] * (pk[a] + kr[a]);
      } else {
#pragma unroll
        for (int a = 0; a < 32; ++a) s += q[a] * pk[a];   // zero-padded K
      }
      sc[pp] = s * 0.17677669529663687f;                  // 1/sqrt(32)
    }
    float mx = sc[0];
#pragma unroll
    for (int pp = 1; pp < 9; ++pp) mx = fmaxf(mx, sc[pp]);
    float den = 0.f;
#pragma unroll
    for (int pp = 0; pp < 9; ++pp) { sc[pp] = __expf(sc[pp] - mx); den += sc[pp]; }
    const float rden = 1.f / den;

    float o[32];
#pragma unroll
    for (int a = 0; a < 32; ++a) o[a] = 0.f;
#pragma unroll
    for (int pp = 0; pp < 9; ++pp) {
      const float wgt = sc[pp] * rden + 1e-8f;
      const int di = pp / 3 - 1, dj = pp % 3 - 1;
      const int hh = h + di, wj = w + dj;
      if ((hh >= 0) && (hh < 64) && (wj >= 0) && (wj < 64)) {
        const float* vr = Vbuf + ((((b << 6) + hh) << 6) + wj) * 128 + head * 32;
#pragma unroll
        for (int a = 0; a < 32; ++a) o[a] += wgt * vr[a];
      }
    }
#pragma unroll
    for (int a = 0; a < 32; ++a)
      att[pix * XB_STR + head * 32 + a] = (bf16_t)o[a];
  }
  __syncthreads();

  // --- final GEMM: att[128x128] @ WfT + bf -> res (LDS) ---
  const int lane  = tid & 31;
  const int wave  = tid >> 5;
  const int mlo   = lane & 15;
  const int khalf = lane >> 4;
  const int mrow  = wave * 16 + mlo;

  v16bf afrag[4];
#pragma unroll
  for (int kc = 0; kc < 4; ++kc)
    afrag[kc] = load_a_frag(att + mrow * XB_STR + kc * 32, khalf);

  const bf16_t* WT = (const bf16_t*)WfTp;
  const int mstore = wave * 16 + khalf * 8;
  for (int nt = 0; nt < 8; ++nt) {
    const int n = nt * 16 + mlo;
    const float bb = bf_[n];
    v8f acc = { bb, bb, bb, bb, bb, bb, bb, bb };
#pragma unroll
    for (int kc = 0; kc < 4; ++kc) {
      v16bf bfrag = *(const v16bf*)(WT + n * 128 + kc * 32 + khalf * 16);
      acc = __builtin_amdgcn_wmma_f32_16x16x32_bf16(
          false, afrag[kc], false, bfrag, (short)0, acc, false, false);
    }
#pragma unroll
    for (int r = 0; r < 8; ++r)
      res[(mstore + r) * XS_STR + n] = acc[r];
  }
  __syncthreads();

  // --- coalesced NCHW store ---
#pragma unroll 4
  for (int i = 0; i < 64; ++i) {
    int idx = tid + i * 256;
    int cout = idx >> 7, pix = idx & 127;
    int p = p0 + pix;
    int h = (p >> 6) & 63, w = p & 63;
    out[((b * 128 + cout) << 12) + (h << 6) + w] = res[pix * XS_STR + cout];
  }
}

// ---------------------------------------------------------------------------
// Launch wrapper
// ---------------------------------------------------------------------------
extern "C" void kernel_launch(void* const* d_in, const int* in_sizes, int n_in,
                              void* d_out, int out_size, void* d_ws, size_t ws_size,
                              hipStream_t stream)
{
  const float* x    = (const float*)d_in[0];
  const float* ln_g = (const float*)d_in[1];
  const float* ln_b = (const float*)d_in[2];
  const float* Wq   = (const float*)d_in[3];
  const float* bq   = (const float*)d_in[4];
  const float* Wk   = (const float*)d_in[5];
  const float* bk   = (const float*)d_in[6];
  const float* Wv   = (const float*)d_in[7];
  const float* bv   = (const float*)d_in[8];
  const float* Wp   = (const float*)d_in[9];
  const float* bp   = (const float*)d_in[10];
  const float* Wf   = (const float*)d_in[11];
  const float* bf_  = (const float*)d_in[12];
  float* out = (float*)d_out;

  char* ws = (char*)d_ws;
  const size_t QKV = (size_t)NPIX * 128 * sizeof(float);        // 16 MB each
  float* Qbuf = (float*)(ws);
  float* Kbuf = (float*)(ws + QKV);
  float* Vbuf = (float*)(ws + 2 * QKV);
  unsigned short* WqT = (unsigned short*)(ws + 3 * QKV);
  unsigned short* WkT = WqT + 128 * 128;
  unsigned short* WvT = WkT + 128 * 128;
  unsigned short* WfT = WvT + 128 * 128;
  float* posk = (float*)(WfT + 128 * 128);                      // 9*128 fp32

  const size_t SMEM_B = (size_t)TILE_M * XS_STR * 4 + (size_t)TILE_M * XB_STR * 2; // 102400
  const size_t SMEM_D = (size_t)TILE_M * XB_STR * 2 + (size_t)TILE_M * XS_STR * 4
                      + 9 * 128 * 4;                                               // 107008

  (void)hipFuncSetAttribute((const void*)ln_qkv_kernel,
                            hipFuncAttributeMaxDynamicSharedMemorySize, (int)SMEM_B);
  (void)hipFuncSetAttribute((const void*)attn_final_kernel,
                            hipFuncAttributeMaxDynamicSharedMemorySize, (int)SMEM_D);

  posk_kernel<<<1, 128, 0, stream>>>(Wp, bp, posk);
  wconv_kernel<<<4, 256, 0, stream>>>(Wq, Wk, Wv, Wf, WqT, WkT, WvT, WfT);
  ln_qkv_kernel<<<NBLK, 256, SMEM_B, stream>>>(x, ln_g, ln_b,
                                               WqT, WkT, WvT, bq, bk, bv,
                                               Qbuf, Kbuf, Vbuf);
  attn_final_kernel<<<NBLK, 256, SMEM_D, stream>>>(Qbuf, Kbuf, Vbuf, posk,
                                                   WfT, bf_, out);
}
